// Linear4Bit_43963285242703
// MI455X (gfx1250) — matrix operand order repaired and checked by reading, not compiled, not run on hardware
//
#include <hip/hip_runtime.h>

// ---------------------------------------------------------------------------
// W4A16 linear for MI455X (gfx1250, wave32, WMMA 16x16x32 f16 -> f32).
//   M=8192 tokens, K=4096, N=11008.  Compute-bound (~0.74 TFLOP vs ~590MB).
//   Block = 256 threads (8 waves), tile 128x128, K-step 64.
//   Software-pipelined: next K-step's global loads (x fp32 + packed int4 W)
//   are held in registers while the current step's WMMAs run from LDS.
//   LDS rows padded to 72 halfs -> conflict-free ds_load_b128 fragment reads.
//   Dequant: nibble -> u16 -> f16 (v_cvt_f16_u16) then packed f16 fma.
// ---------------------------------------------------------------------------

typedef __attribute__((ext_vector_type(16))) _Float16 v16h;
typedef __attribute__((ext_vector_type(8)))  _Float16 v8h;
typedef __attribute__((ext_vector_type(2)))  _Float16 v2h;
typedef __attribute__((ext_vector_type(8)))  float    v8f;
typedef __attribute__((ext_vector_type(4)))  float    v4f;
typedef __attribute__((ext_vector_type(4)))  int      v4i;

#define IN_F  4096
#define OUT_F 11008
#define BM    128
#define BN    128
#define BK    64
#define LDT   72          // padded LDS row stride (halfs); 144B rows, 16B aligned

union Frag { v16h v; v8h h[2]; };

__global__ __launch_bounds__(256)
void linear4bit_wmma(const float* __restrict__ x,
                     const int* __restrict__ wpk,        // one packed byte per int32
                     const _Float16* __restrict__ scales,
                     const _Float16* __restrict__ zeros,
                     const float* __restrict__ bias,
                     float* __restrict__ out) {
  __shared__ _Float16 As[BM * LDT];   // x tile [m][k], f16
  __shared__ _Float16 Bs[BN * LDT];   // W tile [n][k], dequantized f16

  const int tid  = threadIdx.x;
  const int lane = tid & 31;
  const int wave = tid >> 5;
  const int wm   = wave & 1;          // 2 waves along M (64 rows each)
  const int wn   = wave >> 1;         // 4 waves along N (32 cols each)

  const int m0 = blockIdx.y * BM;
  const int n0 = blockIdx.x * BN;

  // staging assignment: thread t -> row t/2 (of both tiles), half-chunk t&1
  const int sr = tid >> 1;            // 0..127
  const int sc = tid & 1;             // 0..1 (32 k-values each)

  // per-output-channel dequant constants, packed for v_pk_fma_f16
  const _Float16 s1 = scales[n0 + sr];
  const _Float16 c1 = (_Float16)(-(float)s1 * (float)zeros[n0 + sr]);  // -s*z
  const v2h dq_s = {s1, s1};
  const v2h dq_c = {c1, c1};

  const int lrow  = lane & 15;
  const int lhalf = lane >> 4;

  const float* xbase = x   + (size_t)(m0 + sr) * IN_F + sc * 32;
  const int*   qbase = wpk + (size_t)(n0 + sr) * (IN_F / 2) + sc * 16;
  _Float16*    ap    = As + sr * LDT + sc * 32;
  _Float16*    bp    = Bs + sr * LDT + sc * 32;

  v8f acc[4][2] = {};                 // 4 (M) x 2 (N) 16x16 f32 accumulators

  // ---- pipeline prologue: load first tile's globals into registers ----
  v4f xa[8];                          // 32 floats of x
  v4i wq[4];                          // 16 packed bytes (one per int32)
#pragma unroll
  for (int i = 0; i < 8; ++i) xa[i] = *(const v4f*)(xbase + i * 4);
#pragma unroll
  for (int i = 0; i < 4; ++i) wq[i] = *(const v4i*)(qbase + i * 4);

  for (int k0 = 0; k0 < IN_F; k0 += BK) {
    // ---- drain staged registers into LDS (convert / dequant) ----
#pragma unroll
    for (int i = 0; i < 4; ++i) {
      v8h h;
      h[0] = (_Float16)xa[2*i][0];   h[1] = (_Float16)xa[2*i][1];
      h[2] = (_Float16)xa[2*i][2];   h[3] = (_Float16)xa[2*i][3];
      h[4] = (_Float16)xa[2*i+1][0]; h[5] = (_Float16)xa[2*i+1][1];
      h[6] = (_Float16)xa[2*i+1][2]; h[7] = (_Float16)xa[2*i+1][3];
      *(v8h*)(ap + i * 8) = h;
    }
#pragma unroll
    for (int i = 0; i < 4; ++i) {
      v8h h;
#pragma unroll
      for (int j = 0; j < 4; ++j) {
        const unsigned int byte = (unsigned int)wq[i][j];
        // u16 -> f16 direct convert (v_cvt_f16_u16), then packed f16 fma
        const unsigned short qlo = (unsigned short)(byte & 15u);
        const unsigned short qhi = (unsigned short)((byte >> 4) & 15u);
        v2h qf = {(_Float16)qlo, (_Float16)qhi};
        v2h w  = qf * dq_s + dq_c;
        h[2 * j]     = w[0];
        h[2 * j + 1] = w[1];
      }
      *(v8h*)(bp + i * 8) = h;
    }
    __syncthreads();

    // ---- issue NEXT tile's global loads; they fly during the WMMAs ----
    if (k0 + BK < IN_F) {
      const float* xp = xbase + (k0 + BK);
      const int*   qp = qbase + ((k0 + BK) >> 1);
#pragma unroll
      for (int i = 0; i < 8; ++i) xa[i] = *(const v4f*)(xp + i * 4);
#pragma unroll
      for (int i = 0; i < 4; ++i) wq[i] = *(const v4i*)(qp + i * 4);
    }

    // ---- compute: 2 k-chunks of 32, 8 WMMAs each per wave ----
#pragma unroll
    for (int kc = 0; kc < 2; ++kc) {
      const int kb = kc * 32;

      // B frag: column N = lane&15; lanes 0-15 -> K 0-7 & 16-23, 16-31 -> K 8-15 & 24-31
      Frag b[2];
#pragma unroll
      for (int nb = 0; nb < 2; ++nb) {
        const _Float16* p = Bs + (wn * 32 + nb * 16 + lrow) * LDT + kb + lhalf * 8;
        b[nb].h[0] = *(const v8h*)p;
        b[nb].h[1] = *(const v8h*)(p + 16);
      }
#pragma unroll
      for (int am = 0; am < 4; ++am) {
        // A frag: row M = lane&15; lanes 0-15 -> K 0-7 & 16-23, 16-31 -> K 8-15 & 24-31
        const _Float16* p = As + (wm * 64 + am * 16 + lrow) * LDT + kb + lhalf * 8;
        Frag a;
        a.h[0] = *(const v8h*)p;
        a.h[1] = *(const v8h*)(p + 16);
#pragma unroll
        for (int nb = 0; nb < 2; ++nb) {
          acc[am][nb] = __builtin_amdgcn_wmma_f32_16x16x32_f16(
              false, a.v, false, b[nb].v, (short)0, acc[am][nb], false, false);
        }
      }
    }
    __syncthreads();
  }

  // ---- epilogue: D layout (lane 0-15: N=lane, VGPR j: M=j; +16: M=j+8) ----
#pragma unroll
  for (int nb = 0; nb < 2; ++nb) {
    const int gn = n0 + wn * 32 + nb * 16 + lrow;
    const float bv = bias[gn];
#pragma unroll
    for (int am = 0; am < 4; ++am) {
      const int gm = m0 + wm * 64 + am * 16 + lhalf * 8;
      float* op = out + (size_t)gm * OUT_F + gn;
#pragma unroll
      for (int j = 0; j < 8; ++j)
        op[(size_t)j * OUT_F] = acc[am][nb][j] + bv;
    }
  }
}

extern "C" void kernel_launch(void* const* d_in, const int* in_sizes, int n_in,
                              void* d_out, int out_size, void* d_ws, size_t ws_size,
                              hipStream_t stream) {
  (void)n_in; (void)out_size; (void)d_ws; (void)ws_size;
  const float*    x      = (const float*)d_in[0];
  const int*      wpk    = (const int*)d_in[1];
  const _Float16* scales = (const _Float16*)d_in[2];
  const _Float16* zeros  = (const _Float16*)d_in[3];
  const float*    bias   = (const float*)d_in[4];
  float*          out    = (float*)d_out;

  const int M = in_sizes[0] / IN_F;           // 8192
  dim3 grid(OUT_F / BN, M / BM);              // (86, 64)
  linear4bit_wmma<<<grid, 256, 0, stream>>>(x, wpk, scales, zeros, bias, out);
}
